// MambaDeepMEns_4827543241067
// MI455X (gfx1250) — matrix operand order repaired and checked by reading, not compiled, not run on hardware
//
#include <hip/hip_runtime.h>
#include <hip/hip_bf16.h>
#include <math.h>

typedef __bf16 bf16_t;
typedef __attribute__((ext_vector_type(16))) __bf16 v16bf;
typedef __attribute__((ext_vector_type(8)))  __bf16 v8bf;
typedef __attribute__((ext_vector_type(8)))  float  v8f;

enum { B_ = 1024, L_ = 23, BLL = B_ * L_ };   // BLL = 23552 = 1472*16

static inline int cdiv(int a, int b) { return (a + b - 1) / b; }

// Load one K-chunk (32 wide) of the A fragment + 4 B fragments.
__device__ __forceinline__
void load_frag(const bf16_t* __restrict__ Aptr, const bf16_t* __restrict__ Wbase,
               const size_t* woff, int k,
               v8bf& alo, v8bf& ahi, v8bf* blo, v8bf* bhi)
{
    alo = *(const v8bf*)(Aptr + k);
    ahi = *(const v8bf*)(Aptr + k + 16);
    #pragma unroll
    for (int j = 0; j < 4; ++j) {
        blo[j] = *(const v8bf*)(Wbase + woff[j] + k);
        bhi[j] = *(const v8bf*)(Wbase + woff[j] + k + 16);
    }
}

// Feed one K-chunk into the 4 accumulators (4 independent v_wmma).
__device__ __forceinline__
void mma4(const v8bf& alo, const v8bf& ahi, const v8bf* blo, const v8bf* bhi, v8f* acc)
{
    v16bf a;
    #pragma unroll
    for (int q = 0; q < 8; ++q) { a[q] = alo[q]; a[q + 8] = ahi[q]; }
    #pragma unroll
    for (int j = 0; j < 4; ++j) {
        v16bf b;
        #pragma unroll
        for (int q = 0; q < 8; ++q) { b[q] = blo[j][q]; b[q + 8] = bhi[j][q]; }
        acc[j] = __builtin_amdgcn_wmma_f32_16x16x32_bf16(
                     false, a, false, b, (short)0, acc[j], false, false);
    }
}

// ---------------------------------------------------------------------------
// WMMA GEMM:  C[M,N] = A[M,K] (bf16, row-major) * W[N,K]^T (bf16) + bias
// One wave32 per 16x64 output tile; 4 accumulators share one A fragment.
// Explicitly double-buffered: loads for chunk k+32 are issued BEFORE the
// WMMAs consuming chunk k, forcing distinct registers so loads overlap WMMA.
// K % 32 == 0, M % 16 == 0. Out-of-range output columns handled by CLAMPING
// the W row (garbage stays in that lane's accumulator, discarded at store),
// so the K-loop has no EXEC divergence and WMMA always runs with EXEC all-1s.
// act: 0 = none, 1 = leaky-relu(0.1). Stores fp32 (Cf) or bf16 (Cb).
// ---------------------------------------------------------------------------
__global__ __launch_bounds__(256)
void k_gemm_bf16(const bf16_t* __restrict__ A,
                 const bf16_t* __restrict__ W,
                 const float*  __restrict__ bias,
                 float* __restrict__ Cf,
                 bf16_t* __restrict__ Cb,
                 int M, int N, int K, int act)
{
    const int lane   = threadIdx.x & 31;
    const int wave   = blockIdx.x * (blockDim.x >> 5) + (threadIdx.x >> 5);
    const int tilesN = (N + 63) >> 6;          // 64-wide N tiles
    const int tiles  = (M >> 4) * tilesN;
    if (wave >= tiles) return;                 // wave-uniform: EXEC stays all-1s

    const int tm   = wave / tilesN;
    const int tn   = wave % tilesN;
    const int half = lane >> 4;                // 0: lanes 0-15, 1: lanes 16-31
    const int l16  = lane & 15;

    // A fragment: ISA 16-bit A 16x32 layout. lane<16 holds K [0,8)+[16,24),
    // lane>=16 holds K [8,16)+[24,32) of row arow.
    const int arow = (tm << 4) + l16;
    const bf16_t* Aptr = A + (size_t)arow * K + (half ? 8 : 0);

    // B fragments: lane holds column wcol[j]; lanes 0-15 K [0,8)+[16,24), 16-31 K [8,16)+[24,32).
    const bf16_t* Wbase = W + (half ? 8 : 0);
    int    wcol[4];
    size_t woff[4];
    #pragma unroll
    for (int j = 0; j < 4; ++j) {
        wcol[j] = (tn << 6) + (j << 4) + l16;
        const int c = (wcol[j] < N) ? wcol[j] : (N - 1);   // clamp, no divergence
        woff[j] = (size_t)c * K;
    }

    v8f acc[4];
    #pragma unroll
    for (int j = 0; j < 4; ++j)
        #pragma unroll
        for (int q = 0; q < 8; ++q) acc[j][q] = 0.0f;

    v8bf a0lo, a0hi, b0lo[4], b0hi[4];
    v8bf a1lo, a1hi, b1lo[4], b1hi[4];
    load_frag(Aptr, Wbase, woff, 0, a0lo, a0hi, b0lo, b0hi);

    for (int k = 0; k < K; k += 64) {
        const bool has2 = (k + 32) < K;        // wave-uniform
        const int  k1   = has2 ? (k + 32) : k;
        load_frag(Aptr, Wbase, woff, k1, a1lo, a1hi, b1lo, b1hi);
        if (k + 96 < K) __builtin_prefetch(Aptr + k + 96, 0, 1);

        mma4(a0lo, a0hi, b0lo, b0hi, acc);     // chunk k (set1 loads in flight)

        const int k2 = ((k + 64) < K) ? (k + 64) : k1;
        load_frag(Aptr, Wbase, woff, k2, a0lo, a0hi, b0lo, b0hi);

        if (has2)
            mma4(a1lo, a1hi, b1lo, b1hi, acc); // chunk k+32 (set0 loads in flight)
    }

    #pragma unroll
    for (int j = 0; j < 4; ++j) {
        if (wcol[j] < N) {
            const float bval = bias ? bias[wcol[j]] : 0.0f;
            #pragma unroll
            for (int q = 0; q < 8; ++q) {
                float v = acc[j][q] + bval;
                if (act == 1) v = (v > 0.0f) ? v : 0.1f * v;
                // C/D layout: VGPR q -> row tm*16 + q (+8 for lanes 16-31)
                const int row = (tm << 4) + q + (half ? 8 : 0);
                const size_t o = (size_t)row * N + wcol[j];
                if (Cf) Cf[o] = v;
                else    Cb[o] = (bf16_t)v;
            }
        }
    }
}

// ---------------------------------------------------------------------------
// Front-end: relu(conv{3,5,7}(x_seq)) concat -> BN -> bf16 c_feat [BLL,192]
// ---------------------------------------------------------------------------
__global__ void k_frontend(const float* __restrict__ xs,
                           const float* __restrict__ w3, const float* __restrict__ b3,
                           const float* __restrict__ w5, const float* __restrict__ b5,
                           const float* __restrict__ w7, const float* __restrict__ b7,
                           const float* __restrict__ g,  const float* __restrict__ bb,
                           const float* __restrict__ mn, const float* __restrict__ vr,
                           bf16_t* __restrict__ out)
{
    int idx = blockIdx.x * blockDim.x + threadIdx.x;
    if (idx >= BLL * 192) return;
    int ch = idx % 192, bl = idx / 192;
    int b = bl / L_, l = bl % L_;

    const float* w; const float* bias; int ks, co;
    if (ch < 64)       { w = w3; bias = b3; ks = 3; co = ch; }
    else if (ch < 128) { w = w5; bias = b5; ks = 5; co = ch - 64; }
    else               { w = w7; bias = b7; ks = 7; co = ch - 128; }
    int pad = ks >> 1;

    float acc = bias[co];
    for (int i = 0; i < 4; ++i)
        for (int j = 0; j < ks; ++j) {
            int t = l + j - pad;
            if (t >= 0 && t < L_)
                acc += w[(co * 4 + i) * ks + j] * xs[((size_t)b * 4 + i) * L_ + t];
        }
    acc = fmaxf(acc, 0.0f);
    float scale = g[ch] * rsqrtf(vr[ch] + 1e-5f);
    out[(size_t)bl * 192 + ch] = (bf16_t)((acc - mn[ch]) * scale + bb[ch]);
}

// xm_flip[b, L-1-l, :] = xm[b, l, :]
__global__ void k_flip(const bf16_t* __restrict__ xm, bf16_t* __restrict__ xf)
{
    int idx = blockIdx.x * blockDim.x + threadIdx.x;
    if (idx >= BLL * 128) return;
    int d = idx % 128, bl = idx / 128;
    int b = bl / L_, l = bl % L_;
    xf[((size_t)(b * L_ + (L_ - 1 - l))) * 128 + d] = xm[(size_t)bl * 128 + d];
}

// pos_emb gather -> bf16 [BLL, 32]
__global__ void k_posgather(const int* __restrict__ pos, const float* __restrict__ emb,
                            bf16_t* __restrict__ out)
{
    int idx = blockIdx.x * blockDim.x + threadIdx.x;
    if (idx >= BLL * 32) return;
    int d = idx % 32, bl = idx / 32;
    int p = pos[bl];
    out[(size_t)bl * 32 + d] = (bf16_t)emb[(size_t)p * 32 + d];
}

// Depthwise causal conv (dc taps) + SiLU on first di columns of xz -> xh bf16
__global__ void k_convsilu(const bf16_t* __restrict__ xz,
                           const float* __restrict__ cw, const float* __restrict__ cb,
                           bf16_t* __restrict__ xh, int di, int dc)
{
    int idx = blockIdx.x * blockDim.x + threadIdx.x;
    if (idx >= BLL * di) return;
    int d = idx % di, bl = idx / di;
    int b = bl / L_, l = bl % L_;
    float acc = cb[d];
    for (int j = 0; j < dc; ++j) {
        int t = l + j - (dc - 1);
        if (t >= 0)
            acc += cw[d * dc + j] * (float)xz[((size_t)(b * L_ + t)) * (2 * di) + d];
    }
    float s = acc / (1.0f + __expf(-acc));
    xh[(size_t)bl * di + d] = (bf16_t)s;
}

// Selective scan, fused with dt_proj+softplus, D skip, SiLU(z) gate.
// One thread per (batch, channel); ds = 16 states in registers.
__global__ void k_scan(const bf16_t* __restrict__ xz, const bf16_t* __restrict__ xh,
                       const float* __restrict__ dbc,
                       const float* __restrict__ dtw, const float* __restrict__ dtb,
                       const float* __restrict__ Alog, const float* __restrict__ Dp,
                       bf16_t* __restrict__ ybuf, int di, int dtr, int nd)
{
    int idx = blockIdx.x * blockDim.x + threadIdx.x;
    if (idx >= B_ * di) return;
    int d = idx % di, b = idx / di;

    float A[16], h[16], wdt[8];
    #pragma unroll
    for (int s = 0; s < 16; ++s) { A[s] = -__expf(Alog[(size_t)d * 16 + s]); h[s] = 0.0f; }
    for (int r = 0; r < dtr; ++r) wdt[r] = dtw[d * dtr + r];
    const float Dd = Dp[d], dtbias = dtb[d];

    for (int t = 0; t < L_; ++t) {
        const size_t base = (size_t)(b * L_ + t);
        const float* drow = dbc + base * nd;

        float dtv = dtbias;
        for (int r = 0; r < dtr; ++r) dtv += drow[r] * wdt[r];
        dtv = (dtv > 20.0f) ? dtv : log1pf(__expf(dtv));   // softplus

        float xhv = (float)xh[base * di + d];
        float dx  = dtv * xhv;
        float y   = 0.0f;
        #pragma unroll
        for (int s = 0; s < 16; ++s) {
            float dA = __expf(dtv * A[s]);
            h[s] = dA * h[s] + dx * drow[dtr + s];          // Bm
            y   += h[s] * drow[dtr + 16 + s];               // Cm
        }
        y += xhv * Dd;
        float z  = (float)xz[base * (2 * di) + di + d];
        float sz = z / (1.0f + __expf(-z));
        ybuf[base * di + d] = (bf16_t)(y * sz);
    }
}

// seq_feat: max over L of concat(out_fwd, out_bwd-on-flipped) -> fused[:, 0:256]
__global__ void k_seqfeat(const float* __restrict__ of, const float* __restrict__ ob,
                          bf16_t* __restrict__ fused)
{
    int idx = blockIdx.x * blockDim.x + threadIdx.x;
    if (idx >= B_ * 256) return;
    int j = idx % 256, b = idx / 256;
    const float* src = (j < 128) ? of : ob;
    int jj = j & 127;
    float m = -1e30f;
    for (int t = 0; t < L_; ++t)
        m = fmaxf(m, src[((size_t)(b * L_ + t)) * 128 + jj]);
    fused[(size_t)b * 320 + j] = (bf16_t)m;
}

// shape branch: bn(relu(conv3(x_shape))); pair-max then max == max over l in [0,22)
__global__ void k_shape(const float* __restrict__ xs,
                        const float* __restrict__ w,  const float* __restrict__ cb,
                        const float* __restrict__ g,  const float* __restrict__ bb,
                        const float* __restrict__ mn, const float* __restrict__ vr,
                        bf16_t* __restrict__ fused)
{
    int idx = blockIdx.x * blockDim.x + threadIdx.x;
    if (idx >= B_ * 32) return;
    int c = idx % 32, b = idx / 32;
    float scale = g[c] * rsqrtf(vr[c] + 1e-5f);
    float best = -1e30f;
    for (int l = 0; l < (L_ / 2) * 2; ++l) {
        float acc = cb[c];
        for (int i = 0; i < 4; ++i)
            for (int j = 0; j < 3; ++j) {
                int t = l + j - 1;
                if (t >= 0 && t < L_)
                    acc += w[(c * 4 + i) * 3 + j] * xs[((size_t)b * 4 + i) * L_ + t];
            }
        acc = fmaxf(acc, 0.0f);
        best = fmaxf(best, (acc - mn[c]) * scale + bb[c]);
    }
    fused[(size_t)b * 320 + 256 + c] = (bf16_t)best;
}

// pos_feat: max over L -> fused[:, 288:320]
__global__ void k_posreduce(const float* __restrict__ op, bf16_t* __restrict__ fused)
{
    int idx = blockIdx.x * blockDim.x + threadIdx.x;
    if (idx >= B_ * 32) return;
    int j = idx % 32, b = idx / 32;
    float m = -1e30f;
    for (int t = 0; t < L_; ++t)
        m = fmaxf(m, op[((size_t)(b * L_ + t)) * 32 + j]);
    fused[(size_t)b * 320 + 288 + j] = (bf16_t)m;
}

// head: sigmoid(h2 @ out_w^T + out_b) -> fp32 output [B_, 1]
__global__ void k_head(const bf16_t* __restrict__ h2, const float* __restrict__ ow,
                       const float* __restrict__ ob, float* __restrict__ out)
{
    int b = blockIdx.x * blockDim.x + threadIdx.x;
    if (b >= B_) return;
    float acc = ob[0];
    for (int j = 0; j < 64; ++j) acc += (float)h2[(size_t)b * 64 + j] * ow[j];
    out[b] = 1.0f / (1.0f + __expf(-acc));
}

// fp32 -> bf16 weight conversion
__global__ void k_cvt(const float* __restrict__ s, bf16_t* __restrict__ d, int n)
{
    int idx = blockIdx.x * blockDim.x + threadIdx.x;
    if (idx < n) d[idx] = (bf16_t)s[idx];
}

// ---------------------------------------------------------------------------
extern "C" void kernel_launch(void* const* d_in, const int* in_sizes, int n_in,
                              void* d_out, int out_size, void* d_ws, size_t ws_size,
                              hipStream_t stream)
{
    (void)in_sizes; (void)n_in; (void)out_size; (void)ws_size;

    const float* x_seq   = (const float*)d_in[0];
    const float* x_shape = (const float*)d_in[1];
    const int*   x_pos   = (const int*)d_in[2];
    int i = 3;
    const float* conv3_w = (const float*)d_in[i++]; const float* conv3_b = (const float*)d_in[i++];
    const float* conv5_w = (const float*)d_in[i++]; const float* conv5_b = (const float*)d_in[i++];
    const float* conv7_w = (const float*)d_in[i++]; const float* conv7_b = (const float*)d_in[i++];
    const float* bnq_g = (const float*)d_in[i++]; const float* bnq_b = (const float*)d_in[i++];
    const float* bnq_m = (const float*)d_in[i++]; const float* bnq_v = (const float*)d_in[i++];
    const float* proj_w = (const float*)d_in[i++]; const float* proj_b = (const float*)d_in[i++];

    struct MP { const float *in_w,*conv_w,*conv_b,*x_w,*dt_w,*dt_b,*A_log,*D,*out_w; };
    auto loadM = [&]() { MP p;
        p.in_w  = (const float*)d_in[i++]; p.conv_w = (const float*)d_in[i++];
        p.conv_b = (const float*)d_in[i++]; p.x_w   = (const float*)d_in[i++];
        p.dt_w  = (const float*)d_in[i++]; p.dt_b  = (const float*)d_in[i++];
        p.A_log = (const float*)d_in[i++]; p.D     = (const float*)d_in[i++];
        p.out_w = (const float*)d_in[i++]; return p; };
    MP mf = loadM(); MP mb = loadM();
    const float* convs_w = (const float*)d_in[i++]; const float* convs_b = (const float*)d_in[i++];
    const float* bns_g = (const float*)d_in[i++]; const float* bns_b = (const float*)d_in[i++];
    const float* bns_m = (const float*)d_in[i++]; const float* bns_v = (const float*)d_in[i++];
    const float* pos_emb = (const float*)d_in[i++];
    MP mp = loadM();
    const float* fc1_w = (const float*)d_in[i++]; const float* fc1_b = (const float*)d_in[i++];
    const float* fc2_w = (const float*)d_in[i++]; const float* fc2_b = (const float*)d_in[i++];
    const float* outw  = (const float*)d_in[i++]; const float* outb  = (const float*)d_in[i++];

    // ---- carve workspace ----
    char* ws = (char*)d_ws; size_t off = 0;
    auto carve = [&](size_t bytes) -> char* {
        off = (off + 255) & ~(size_t)255;
        char* p = ws + off; off += bytes; return p; };

    bf16_t* w_proj = (bf16_t*)carve(128 * 192 * 2);
    bf16_t* wf_in  = (bf16_t*)carve(512 * 128 * 2);
    bf16_t* wf_x   = (bf16_t*)carve(40 * 256 * 2);
    bf16_t* wf_o   = (bf16_t*)carve(128 * 256 * 2);
    bf16_t* wb_in  = (bf16_t*)carve(512 * 128 * 2);
    bf16_t* wb_x   = (bf16_t*)carve(40 * 256 * 2);
    bf16_t* wb_o   = (bf16_t*)carve(128 * 256 * 2);
    bf16_t* wp_in  = (bf16_t*)carve(128 * 32 * 2);
    bf16_t* wp_x   = (bf16_t*)carve(34 * 64 * 2);
    bf16_t* wp_o   = (bf16_t*)carve(32 * 64 * 2);
    bf16_t* w_fc1  = (bf16_t*)carve(128 * 320 * 2);
    bf16_t* w_fc2  = (bf16_t*)carve(64 * 128 * 2);

    bf16_t* c_feat  = (bf16_t*)carve((size_t)BLL * 192 * 2);
    bf16_t* xm      = (bf16_t*)carve((size_t)BLL * 128 * 2);
    bf16_t* xm_flip = (bf16_t*)carve((size_t)BLL * 128 * 2);
    bf16_t* embp    = (bf16_t*)carve((size_t)BLL * 32 * 2);
    bf16_t* xz      = (bf16_t*)carve((size_t)BLL * 512 * 2);   // reused per mamba
    bf16_t* xh      = (bf16_t*)carve((size_t)BLL * 256 * 2);
    float*  dbc     = (float*)carve((size_t)BLL * 40 * 4);
    bf16_t* ybuf    = (bf16_t*)carve((size_t)BLL * 256 * 2);
    float*  out_f   = (float*)carve((size_t)BLL * 128 * 4);
    float*  out_bw  = (float*)carve((size_t)BLL * 128 * 4);
    float*  out_p   = (float*)carve((size_t)BLL * 32 * 4);
    bf16_t* fused   = (bf16_t*)carve((size_t)B_ * 320 * 2);
    bf16_t* h1      = (bf16_t*)carve((size_t)B_ * 128 * 2);
    bf16_t* h2      = (bf16_t*)carve((size_t)B_ * 64 * 2);

    auto cvt = [&](const float* s, bf16_t* d, int n) {
        k_cvt<<<cdiv(n, 256), 256, 0, stream>>>(s, d, n); };
    auto gemm = [&](const bf16_t* A, const bf16_t* W, const float* bias,
                    float* Cf, bf16_t* Cb, int M, int N, int K, int act) {
        int tiles = (M / 16) * cdiv(N, 64);
        k_gemm_bf16<<<cdiv(tiles, 8), 256, 0, stream>>>(A, W, bias, Cf, Cb, M, N, K, act); };

    // ---- bf16 weight conversion ----
    cvt(proj_w, w_proj, 128 * 192);
    cvt(mf.in_w, wf_in, 512 * 128); cvt(mf.x_w, wf_x, 40 * 256); cvt(mf.out_w, wf_o, 128 * 256);
    cvt(mb.in_w, wb_in, 512 * 128); cvt(mb.x_w, wb_x, 40 * 256); cvt(mb.out_w, wb_o, 128 * 256);
    cvt(mp.in_w, wp_in, 128 * 32);  cvt(mp.x_w, wp_x, 34 * 64);  cvt(mp.out_w, wp_o, 32 * 64);
    cvt(fc1_w, w_fc1, 128 * 320);   cvt(fc2_w, w_fc2, 64 * 128);

    // ---- front-end + projection ----
    k_frontend<<<cdiv(BLL * 192, 256), 256, 0, stream>>>(
        x_seq, conv3_w, conv3_b, conv5_w, conv5_b, conv7_w, conv7_b,
        bnq_g, bnq_b, bnq_m, bnq_v, c_feat);
    gemm(c_feat, w_proj, proj_b, nullptr, xm, BLL, 128, 192, 0);
    k_flip<<<cdiv(BLL * 128, 256), 256, 0, stream>>>(xm, xm_flip);
    k_posgather<<<cdiv(BLL * 32, 256), 256, 0, stream>>>(x_pos, pos_emb, embp);

    // ---- mamba blocks ----
    auto run_mamba = [&](const bf16_t* x, const MP& p,
                         const bf16_t* wIn, const bf16_t* wX, const bf16_t* wO,
                         int dmodel, int di, int dtr, int dc, float* outbuf) {
        int nd = dtr + 32;
        gemm(x, wIn, nullptr, nullptr, xz, BLL, 2 * di, dmodel, 0);
        k_convsilu<<<cdiv(BLL * di, 256), 256, 0, stream>>>(xz, p.conv_w, p.conv_b, xh, di, dc);
        gemm(xh, wX, nullptr, dbc, nullptr, BLL, nd, di, 0);
        k_scan<<<cdiv(B_ * di, 256), 256, 0, stream>>>(
            xz, xh, dbc, p.dt_w, p.dt_b, p.A_log, p.D, ybuf, di, dtr, nd);
        gemm(ybuf, wO, nullptr, outbuf, nullptr, BLL, dmodel, di, 0);
    };
    run_mamba(xm,      mf, wf_in, wf_x, wf_o, 128, 256, 8, 4, out_f);
    run_mamba(xm_flip, mb, wb_in, wb_x, wb_o, 128, 256, 8, 4, out_bw);
    run_mamba(embp,    mp, wp_in, wp_x, wp_o, 32,  64,  2, 2, out_p);

    // ---- feature fusion ----
    k_seqfeat<<<cdiv(B_ * 256, 256), 256, 0, stream>>>(out_f, out_bw, fused);
    k_shape<<<cdiv(B_ * 32, 256), 256, 0, stream>>>(
        x_shape, convs_w, convs_b, bns_g, bns_b, bns_m, bns_v, fused);
    k_posreduce<<<cdiv(B_ * 32, 256), 256, 0, stream>>>(out_p, fused);

    // ---- MLP head ----
    gemm(fused, w_fc1, fc1_b, nullptr, h1, B_, 128, 320, 1);
    gemm(h1, w_fc2, fc2_b, nullptr, h2, B_, 64, 128, 1);
    k_head<<<cdiv(B_, 256), 256, 0, stream>>>(h2, outw, outb, (float*)d_out);
}
